// BertSAE_3779571221061
// MI455X (gfx1250) — matrix operand order repaired and checked by reading, not compiled, not run on hardware
//
#include <hip/hip_runtime.h>

// ---------------------------------------------------------------------------
// Top-K Sparse Autoencoder forward, fused for MI455X (gfx1250, wave32, WMMA).
//   encode:  pre = x @ W_enc^T + b_enc      [8192, 24576]  (309 GFLOP)
//   topk:    per-row top-32 (values+indices)
//   decode:  recon = scatter(top) @ W_dec^T + b_dec        [8192, 768]
// pre_acts never touches HBM: each WG computes a 128-row x 3072-col strip
// with v_wmma_f32_16x16x32_bf16 and folds top-k per 128x128 LDS tile.
// Loop nest: k outer, ms inner with 8 independent accumulators -> B loaded
// once per k-step (8x less L2 traffic) and 8 independent WMMA chains.
// ---------------------------------------------------------------------------

typedef __attribute__((ext_vector_type(16))) __bf16 v16bf;
typedef __attribute__((ext_vector_type(8)))  __bf16 v8bf;
typedef __attribute__((ext_vector_type(8)))  float  v8f;

#define N_ROWS   8192
#define D_MODEL  768
#define D_SAE    24576
#define TOPK     32

#define TILE_M   128                 // rows per workgroup
#define MSUB     (TILE_M / 16)       // 8 row sub-tiles (= independent acc chains)
#define S_STRIPS 8                   // column strips (partial top-k per strip)
#define S_STRIP  (D_SAE / S_STRIPS)  // 3072
#define TILE_S   128                 // cols per iteration (8 waves * 16)
#define ITERS    (S_STRIP / TILE_S)  // 24
#define XSTR     (D_MODEL + 8)       // bf16 elems; padded LDS row stride
#define TSTR     (TILE_S + 1)        // padded f32 tile stride (bank spread)

#define XS_BYTES   (TILE_M * XSTR * 2)   // 198656
#define TILE_BYTES (TILE_M * TSTR * 4)   // 66048
#define HV_BYTES   (TILE_M * TOPK * 4)   // 16384
#define HI_BYTES   (TILE_M * TOPK * 4)   // 16384
#define ENC_SMEM   (XS_BYTES + TILE_BYTES + HV_BYTES + HI_BYTES)  // 297472 < 320K

// ---------------- W_enc fp32 -> bf16 (halves the dominant L2 stream) -------
__global__ void __launch_bounds__(256)
k_cvt_wenc(const float* __restrict__ w, __bf16* __restrict__ out) {
    const size_t i = ((size_t)blockIdx.x * 256 + threadIdx.x) * 4;
    const float4 v = *(const float4*)(w + i);
    union { __bf16 b[4]; unsigned long long u; } pk;
    pk.b[0] = (__bf16)v.x; pk.b[1] = (__bf16)v.y;
    pk.b[2] = (__bf16)v.z; pk.b[3] = (__bf16)v.w;
    *(unsigned long long*)(out + i) = pk.u;
}

// ---------------- W_dec [768][24576] -> W_decT [24576][768] ----------------
__global__ void __launch_bounds__(256)
k_transpose_wdec(const float* __restrict__ wdec, float* __restrict__ wdecT) {
    __shared__ float t[32][33];
    const int tx = threadIdx.x, ty = threadIdx.y;     // block (32, 8)
    const int s0 = blockIdx.x * 32, d0 = blockIdx.y * 32;
#pragma unroll
    for (int i = 0; i < 32; i += 8)
        t[ty + i][tx] = wdec[(size_t)(d0 + ty + i) * D_SAE + s0 + tx];
    __syncthreads();
#pragma unroll
    for (int i = 0; i < 32; i += 8)
        wdecT[(size_t)(s0 + ty + i) * D_MODEL + d0 + tx] = t[tx][ty + i];
}

// ---------------- fused encode GEMM (WMMA bf16) + per-strip top-32 ---------
__global__ void __launch_bounds__(256)
k_encode_topk(const float* __restrict__ x, const __bf16* __restrict__ wenc,
              const float* __restrict__ b_enc,
              float* __restrict__ pv, int* __restrict__ pi) {
    extern __shared__ char smem[];
    __bf16* xs  = (__bf16*)smem;                                   // A strip, bf16
    float* tile = (float*)(smem + XS_BYTES);                       // 128x128 pre_acts
    float* hv   = (float*)(smem + XS_BYTES + TILE_BYTES);          // heaps: values
    int*   hi   = (int*)(smem + XS_BYTES + TILE_BYTES + HV_BYTES); // heaps: indices

    const int tid   = threadIdx.x;
    const int wave  = tid >> 5;
    const int lane  = tid & 31;
    const int strip = blockIdx.x & (S_STRIPS - 1);
    const int row0  = (blockIdx.x / S_STRIPS) * TILE_M;
    const int sbase = strip * S_STRIP;

    // Stage x[row0 .. row0+127][0..767] into LDS as bf16 (read once per WG).
    for (int i = tid; i < TILE_M * D_MODEL / 4; i += 256) {
        const int e = i * 4, r = e / D_MODEL, c = e % D_MODEL;
        const float4 v = *(const float4*)(x + (size_t)(row0 + r) * D_MODEL + c);
        union { __bf16 b[4]; unsigned long long u; } pk;
        pk.b[0] = (__bf16)v.x; pk.b[1] = (__bf16)v.y;
        pk.b[2] = (__bf16)v.z; pk.b[3] = (__bf16)v.w;
        *(unsigned long long*)(xs + r * XSTR + c) = pk.u;
    }
    float hmin = -__builtin_inff();
    int   hargmin = 0;
    if (tid < TILE_M) {
#pragma unroll
        for (int j = 0; j < TOPK; ++j) {
            hv[tid * TOPK + j] = -__builtin_inff();
            hi[tid * TOPK + j] = 0;
        }
    }
    __syncthreads();

    const int n    = lane & 15;    // WMMA: A row / B col / C col for this lane
    const int half = lane >> 4;    // K half-select per ISA 16-bit layouts
    const int wcol = wave * 16;

    for (int it = 0; it < ITERS; ++it) {
        const int col0 = sbase + it * TILE_S + wcol;
        const float be = b_enc[col0 + n];
        // B fragment base: lane n holds W_enc[col0+n][k0 + half*16 .. +15]
        const __bf16* wbase = wenc + (size_t)(col0 + n) * D_MODEL + half * 16;
        __builtin_prefetch(wbase + (size_t)TILE_S * D_MODEL, 0, 1);  // next tile

        v8f acc[MSUB];
#pragma unroll
        for (int ms = 0; ms < MSUB; ++ms)
            acc[ms] = (v8f){0.f, 0.f, 0.f, 0.f, 0.f, 0.f, 0.f, 0.f};

        // A fragment base: lanes 0-15 hold K {0..15}; lanes 16-31 {16..31}
        const __bf16* abase = xs + n * XSTR + half * 16;

#pragma unroll 2
        for (int k0 = 0; k0 < D_MODEL; k0 += 32) {
            const v16bf b = *(const v16bf*)(wbase + k0);   // one B load per k-step
#pragma unroll
            for (int ms = 0; ms < MSUB; ++ms) {
                const __bf16* arow = abase + ms * 16 * XSTR + k0;
                union { v16bf v; v8bf h[2]; } a;
                a.h[0] = *(const v8bf*)(arow);             // ds_load_b128
                a.h[1] = *(const v8bf*)(arow + 8);         // ds_load_b128
                acc[ms] = __builtin_amdgcn_wmma_f32_16x16x32_bf16(
                              false, a.v, false, b, (short)0, acc[ms], false, false);
            }
        }

        // C layout: VGPR j -> M = j (lanes 0-15) or M = j+8 (lanes 16-31)
        const int mbofs = half << 3;
#pragma unroll
        for (int ms = 0; ms < MSUB; ++ms) {
            const int mb = ms * 16 + mbofs;
#pragma unroll
            for (int j = 0; j < 8; ++j)
                tile[(mb + j) * TSTR + wcol + n] = acc[ms][j] + be;
        }
        __syncthreads();

        // Threshold-filtered top-32 insert: thread r owns row r.
        if (tid < TILE_M) {
            const float* trow = tile + tid * TSTR;
            const int ibase = sbase + it * TILE_S;
            float* mhv = hv + tid * TOPK;
            int*   mhi = hi + tid * TOPK;
            for (int cc = 0; cc < TILE_S; ++cc) {
                const float v = trow[cc];
                if (v > hmin) {
                    mhv[hargmin] = v; mhi[hargmin] = ibase + cc;
                    float m = mhv[0]; int am = 0;
#pragma unroll
                    for (int j = 1; j < TOPK; ++j) {
                        const float t = mhv[j];
                        if (t < m) { m = t; am = j; }
                    }
                    hmin = m; hargmin = am;
                }
            }
        }
        __syncthreads();
    }

    if (tid < TILE_M) {
        const size_t pb = (((size_t)(row0 + tid)) * S_STRIPS + strip) * TOPK;
#pragma unroll
        for (int j = 0; j < TOPK; ++j) {
            pv[pb + j] = hv[tid * TOPK + j];
            pi[pb + j] = hi[tid * TOPK + j];
        }
    }
}

// ---------------- merge 8 strip-partials -> sorted global top-32 -----------
__global__ void __launch_bounds__(256)
k_merge_topk(const float* __restrict__ pv, const int* __restrict__ pi,
             float* __restrict__ tv, int* __restrict__ ti) {
    extern __shared__ char smem[];
    float* hv = (float*)smem;                     // 256*32 f32
    int*   hi = (int*)(smem + 256 * TOPK * 4);    // 256*32 i32
    const int t   = threadIdx.x;
    const int row = blockIdx.x * 256 + t;
    float* mv = hv + t * TOPK;
    int*   mi = hi + t * TOPK;
    float hmin = -__builtin_inff();
    int   am = 0;
#pragma unroll
    for (int j = 0; j < TOPK; ++j) { mv[j] = -__builtin_inff(); mi[j] = 0; }

    const float* rv = pv + (size_t)row * (S_STRIPS * TOPK);
    const int*   ri = pi + (size_t)row * (S_STRIPS * TOPK);
    for (int c = 0; c < S_STRIPS * TOPK; ++c) {
        const float v = rv[c];
        if (v > hmin) {
            mv[am] = v; mi[am] = ri[c];
            float m = mv[0]; int a = 0;
#pragma unroll
            for (int j = 1; j < TOPK; ++j) {
                const float q = mv[j];
                if (q < m) { m = q; a = j; }
            }
            hmin = m; am = a;
        }
    }
    // selection sort, descending (jax top_k returns sorted values)
    for (int a = 0; a < TOPK; ++a) {
        int best = a; float bm = mv[a];
        for (int j = a + 1; j < TOPK; ++j) {
            const float q = mv[j];
            if (q > bm) { bm = q; best = j; }
        }
        const float fv = mv[a]; const int fi = mi[a];
        mv[a] = mv[best]; mi[a] = mi[best];
        mv[best] = fv;    mi[best] = fi;
        tv[(size_t)row * TOPK + a] = mv[a];
        ti[(size_t)row * TOPK + a] = mi[a];
    }
}

// ---------------- sparse decode: recon = top @ W_decT rows + b_dec ---------
__global__ void __launch_bounds__(256)
k_decode(const float* __restrict__ tv, const int* __restrict__ ti,
         const float* __restrict__ wdecT, const float* __restrict__ b_dec,
         float* __restrict__ recon) {
    __shared__ float av[TOPK];
    __shared__ int   ai[TOPK];
    const int row = blockIdx.x, t = threadIdx.x;
    if (t < TOPK) {
        av[t] = tv[(size_t)row * TOPK + t];
        ai[t] = ti[(size_t)row * TOPK + t];
    }
    __syncthreads();
    float a0 = b_dec[t], a1 = b_dec[t + 256], a2 = b_dec[t + 512];
#pragma unroll 4
    for (int k = 0; k < TOPK; ++k) {
        const float s = av[k];
        const float* wr = wdecT + (size_t)ai[k] * D_MODEL;   // coalesced row
        a0 += s * wr[t]; a1 += s * wr[t + 256]; a2 += s * wr[t + 512];
    }
    float* out = recon + (size_t)row * D_MODEL;
    out[t] = a0; out[t + 256] = a1; out[t + 512] = a2;
}

// ---------------------------------------------------------------------------
extern "C" void kernel_launch(void* const* d_in, const int* in_sizes, int n_in,
                              void* d_out, int out_size, void* d_ws, size_t ws_size,
                              hipStream_t stream) {
    (void)in_sizes; (void)n_in; (void)out_size; (void)ws_size;
    const float* x     = (const float*)d_in[0];
    const float* W_enc = (const float*)d_in[1];
    const float* b_enc = (const float*)d_in[2];
    const float* W_dec = (const float*)d_in[3];
    const float* b_dec = (const float*)d_in[4];

    // d_out: recon [8192*768] f32 | top_acts [8192*32] f32 | top_idx [8192*32] i32
    float* recon = (float*)d_out;
    float* top_v = recon + (size_t)N_ROWS * D_MODEL;
    int*   top_i = (int*)(top_v + (size_t)N_ROWS * TOPK);

    // workspace carve (~130 MB): W_enc-bf16 | W_decT f32 | strip partial top-k
    char* ws = (char*)d_ws;
    __bf16* wencb = (__bf16*)ws;
    size_t off = (size_t)D_SAE * D_MODEL * 2;
    float* wdecT = (float*)(ws + off);
    off += (size_t)D_SAE * D_MODEL * 4;
    float* pv = (float*)(ws + off);
    off += (size_t)N_ROWS * S_STRIPS * TOPK * 4;
    int* pi = (int*)(ws + off);

    k_cvt_wenc<<<(D_SAE * D_MODEL) / 1024, 256, 0, stream>>>(W_enc, wencb);
    k_transpose_wdec<<<dim3(D_SAE / 32, D_MODEL / 32), dim3(32, 8), 0, stream>>>(W_dec, wdecT);
    k_encode_topk<<<(N_ROWS / TILE_M) * S_STRIPS, 256, ENC_SMEM, stream>>>(x, wencb, b_enc, pv, pi);
    k_merge_topk<<<N_ROWS / 256, 256, 256 * TOPK * 8, stream>>>(pv, pi, top_v, top_i);
    k_decode<<<N_ROWS, 256, 0, stream>>>(top_v, top_i, wdecT, b_dec, recon);
}
